// CrossAttention_29283087025000
// MI455X (gfx1250) — matrix operand order repaired and checked by reading, not compiled, not run on hardware
//
#include <hip/hip_runtime.h>

typedef __attribute__((ext_vector_type(16))) _Float16 v16h;
typedef __attribute__((ext_vector_type(8)))  float    v8f;
typedef __attribute__((ext_vector_type(4)))  int      vi4;
typedef _Float16 half_t;

#define BATCH 2
#define QLEN 256
#define KLEN 2048
#define DIM 256
#define HEADS 8
#define HD 32

#if defined(__has_builtin)
#if __has_builtin(__builtin_amdgcn_global_load_async_to_lds_b128) && \
    __has_builtin(__builtin_amdgcn_s_wait_asynccnt)
#define USE_ASYNC_LDS 1
#endif
#endif
#ifndef USE_ASYNC_LDS
#define USE_ASYNC_LDS 0
#endif

#if USE_ASYNC_LDS
// Probed signature: (vi4* src_generic, AS3 vi4* lds_dst, imm offset, imm cpol)
#define AS_GLOBAL(p) ((vi4*)(void*)(p))
#define AS_LDS(p)                                            \
  ((__attribute__((address_space(3))) vi4*)(                 \
      (__attribute__((address_space(3))) void*)(void*)(p)))
#endif

__device__ __forceinline__ v8f wmma_f16(v16h a, v16h b, v8f c) {
  return __builtin_amdgcn_wmma_f32_16x16x32_f16(false, a, false, b, (short)0, c,
                                                false, false);
}

// A fragment (16x32 f16) from row-major fp32 [M,K]: lane L -> row m0+(L&15),
// K elements {kb..kb+15}, kb = (L>=16) ? 16 : 0 (contiguous).
__device__ __forceinline__ v16h loadA_f32(const float* A, int lda, int m0, int k0) {
  int lane = threadIdx.x & 31;
  const float* p = A + (size_t)(m0 + (lane & 15)) * lda + k0 + ((lane & 16) ? 16 : 0);
  v16h r;
#pragma unroll
  for (int e = 0; e < 8; ++e) r[e] = (_Float16)p[e];
#pragma unroll
  for (int e = 0; e < 8; ++e) r[8 + e] = (_Float16)p[8 + e];
  return r;
}

// Same, f16 source.
__device__ __forceinline__ v16h loadA_f16(const half_t* A, int lda, int m0, int k0) {
  int lane = threadIdx.x & 31;
  const half_t* p = A + (size_t)(m0 + (lane & 15)) * lda + k0 + ((lane & 16) ? 16 : 0);
  v16h r;
#pragma unroll
  for (int e = 0; e < 8; ++e) r[e] = p[e];
#pragma unroll
  for (int e = 0; e < 8; ++e) r[8 + e] = p[8 + e];
  return r;
}

// B fragment (32x16) where B[k][n] = W[n][k] (W row-major [N,K], fp32):
// lane L -> col n0+(L&15), K = kb16 + e, kb16 = (L>=16) ? 16 : 0 (contiguous).
__device__ __forceinline__ v16h loadB_NK_f32(const float* W, int ldw, int n0, int k0) {
  int lane = threadIdx.x & 31;
  const float* p = W + (size_t)(n0 + (lane & 15)) * ldw + k0 + ((lane & 16) ? 16 : 0);
  v16h r;
#pragma unroll
  for (int e = 0; e < 16; ++e) r[e] = (_Float16)p[e];
  return r;
}

// ---------------- Kernel 1: Q projection ----------------
// q16[b,h,q,hd] = f16(entities @ Wq^T + bq), M=512 N=256 K=256
__global__ void proj_q_kernel(const float* __restrict__ ent, const float* __restrict__ Wq,
                              const float* __restrict__ bq, half_t* __restrict__ qout) {
  int wid = threadIdx.x >> 5, lane = threadIdx.x & 31;
  int tile = blockIdx.x * 4 + wid;  // 512 tiles = 32 x 16
  int mt = tile >> 4, nt = tile & 15;
  v8f acc = {};
#pragma unroll
  for (int kc = 0; kc < DIM; kc += 32) {
    v16h a = loadA_f32(ent, DIM, mt * 16, kc);
    v16h w = loadB_NK_f32(Wq, DIM, nt * 16, kc);
    acc = wmma_f16(a, w, acc);
  }
  int n = nt * 16 + (lane & 15);
  int hh = n >> 5, d = n & 31;
  float bias = bq[n];
  int rb = (lane & 16) ? 8 : 0;
#pragma unroll
  for (int e = 0; e < 8; ++e) {
    int m = mt * 16 + rb + e;
    int b = m >> 8, qr = m & 255;
    qout[(((size_t)(b * HEADS + hh) * QLEN) + qr) * HD + d] = (_Float16)(acc[e] + bias);
  }
}

// ---------------- Kernel 2: KV projection ----------------
// M=4096 N=512 K=256; n<256 -> k16[b,h,kpos,hd]; else vT16[b,h,d,kpos]
__global__ void proj_kv_kernel(const float* __restrict__ img, const float* __restrict__ Wkv,
                               const float* __restrict__ bkv, half_t* __restrict__ kout,
                               half_t* __restrict__ voutT) {
  int wid = threadIdx.x >> 5, lane = threadIdx.x & 31;
  int tile = blockIdx.x * 4 + wid;  // 8192 tiles = 256 x 32
  int mt = tile >> 5, nt = tile & 31;
  v8f acc = {};
#pragma unroll
  for (int kc = 0; kc < DIM; kc += 32) {
    v16h a = loadA_f32(img, DIM, mt * 16, kc);
    v16h w = loadB_NK_f32(Wkv, DIM, nt * 16, kc);
    acc = wmma_f16(a, w, acc);
  }
  int n = nt * 16 + (lane & 15);
  float bias = bkv[n];
  int rb = (lane & 16) ? 8 : 0;
#pragma unroll
  for (int e = 0; e < 8; ++e) {
    int m = mt * 16 + rb + e;
    int b = m >> 11, kp = m & 2047;
    float val = acc[e] + bias;
    if (n < DIM) {
      int hh = n >> 5, d = n & 31;
      kout[(((size_t)(b * HEADS + hh) * KLEN) + kp) * HD + d] = (_Float16)val;
    } else {
      int n2 = n - DIM;
      int hh = n2 >> 5, d = n2 & 31;
      voutT[(((size_t)(b * HEADS + hh) * HD) + d) * KLEN + kp] = (_Float16)val;
    }
  }
}

// ---------------- Kernel 3: continuous position bias ----------------
// cpb[b,h,q,kk] = relu(rd @ W1^T + b1) @ W2^T + b2, one 16-point tile per wave.
// Hidden layer (K=2) on VALU, built directly in WMMA A-fragment layout; second
// layer (K=256, N=8 padded to 16) as 8 chained WMMAs.
__global__ void cpb_kernel(const float* __restrict__ rd, const float* __restrict__ W1,
                           const float* __restrict__ b1, const float* __restrict__ W2,
                           const float* __restrict__ b2, half_t* __restrict__ cpbout) {
  __shared__ float sW1a[256], sW1b[256], sB1[256];
  __shared__ _Float16 sW2[8 * 256];
  int tid = threadIdx.x;
  sW1a[tid] = W1[2 * tid];
  sW1b[tid] = W1[2 * tid + 1];
  sB1[tid] = b1[tid];
#pragma unroll
  for (int hh = 0; hh < 8; ++hh) sW2[hh * 256 + tid] = (_Float16)W2[hh * 256 + tid];
  __syncthreads();

  int wid = tid >> 5, lane = tid & 31;
  int t = blockIdx.x * 8 + wid;  // 65536 tiles of 16 points
  int p0 = t * 16;
  int b = p0 >> 19;        // / (Q*K)
  int rem = p0 & 524287;
  int q = rem >> 11;       // / K
  int kk = rem & 2047;
  int row = lane & 15;
  size_t pidx = ((size_t)((b * QLEN + q)) * KLEN + kk + row) * 2;
  float d0 = rd[pidx], d1 = rd[pidx + 1];
  int kbA = (lane & 16) ? 16 : 0;
  int col = lane & 15;
  int kb16 = (lane & 16) ? 16 : 0;
  v8f acc = {};
#pragma unroll
  for (int c = 0; c < 8; ++c) {
    v16h hA;
#pragma unroll
    for (int e = 0; e < 8; ++e) {
      int j = 32 * c + kbA + e;
      hA[e] = (_Float16)fmaxf(fmaf(d0, sW1a[j], fmaf(d1, sW1b[j], sB1[j])), 0.f);
      int j2 = j + 8;
      hA[8 + e] = (_Float16)fmaxf(fmaf(d0, sW1a[j2], fmaf(d1, sW1b[j2], sB1[j2])), 0.f);
    }
    v16h w2B;
#pragma unroll
    for (int e = 0; e < 16; ++e)
      w2B[e] = (col < 8) ? sW2[col * 256 + 32 * c + kb16 + e] : (_Float16)0.f;
    acc = wmma_f16(hA, w2B, acc);
  }
  if (col < 8) {
    float bias = b2[col];
    int rb = (lane & 16) ? 8 : 0;
#pragma unroll
    for (int e = 0; e < 8; ++e) {
      size_t o = (((size_t)(b * HEADS + col) * QLEN + q) * KLEN) + kk + rb + e;
      cpbout[o] = (_Float16)(acc[e] + bias);
    }
  }
}

// ---------------- Kernel 4: fused flash attention ----------------
// Block = 4 waves sharing one (b,h); each wave owns a 16-row q tile. K/V
// 32-chunks staged in LDS via async global->LDS copies (ASYNCcnt path),
// S = WMMA(q,k^T)*scale + cpb, online softmax, P transposed through LDS,
// O += WMMA(P, V).
__global__ void attn_kernel(const half_t* __restrict__ qb, const half_t* __restrict__ kb,
                            const half_t* __restrict__ vbT, const half_t* __restrict__ cpb,
                            half_t* __restrict__ att) {
  __shared__ __align__(16) _Float16 sk[32][32];       // [kpos_local][d]
  __shared__ __align__(16) _Float16 sv[32][32];       // [d][kpos_local]
  __shared__ __align__(16) _Float16 pbuf[4][16][32];  // per-wave P transpose
  int tid = threadIdx.x;
  int wid = tid >> 5, lane = tid & 31;
  int bh = blockIdx.x >> 2;  // 16 (b,h) pairs
  int qg = blockIdx.x & 3;
  int qt = qg * 4 + wid;     // q tile 0..15
  int b = bh >> 3, h = bh & 7;
  const half_t* qp = qb + (size_t)bh * QLEN * HD;
  const half_t* kp = kb + (size_t)bh * KLEN * HD;
  const half_t* vpT = vbT + (size_t)bh * HD * KLEN;
  const half_t* cp = cpb + (size_t)bh * QLEN * KLEN;

  v16h qA = loadA_f16(qp, HD, qt * 16, 0);
  v8f O0 = {}, O1 = {};
  float mi[8], li[8];
#pragma unroll
  for (int e = 0; e < 8; ++e) { mi[e] = -1e30f; li[e] = 0.f; }
  int rb = (lane & 16) ? 8 : 0;
  int c0 = lane & 15;
  int kb16 = (lane & 16) ? 16 : 0;
  const float scale = 0.17677669529663687f;  // 32^-0.5

  for (int kc = 0; kc < KLEN; kc += 32) {
    // ---- cooperative K/V chunk staging: 128 threads x 16B per tile ----
    {
      const char* gK = (const char*)(kp + (size_t)kc * HD) + tid * 16;  // contiguous 2KB
      char* lK = (char*)(&sk[0][0]) + tid * 16;
      int dd = tid >> 2, seg = tid & 3;  // vT rows: 64B each, 4 segments
      const char* gV = (const char*)(vpT + (size_t)dd * KLEN + kc) + seg * 16;
      char* lV = (char*)(&sv[dd][0]) + seg * 16;
#if USE_ASYNC_LDS
      __builtin_amdgcn_global_load_async_to_lds_b128(AS_GLOBAL(gK), AS_LDS(lK), 0, 0);
      __builtin_amdgcn_global_load_async_to_lds_b128(AS_GLOBAL(gV), AS_LDS(lV), 0, 0);
      __builtin_amdgcn_s_wait_asynccnt(0);
#else
      *(uint4*)lK = *(const uint4*)gK;
      *(uint4*)lV = *(const uint4*)gV;
#endif
    }
    __syncthreads();

    // ---- fragments from LDS (all contiguous 16-half reads) ----
    v16h kB0, kB1, vB0, vB1;
    {
      const _Float16* pk0 = &sk[c0][kb16];        // B[d][kpos], cols kc..kc+15
      const _Float16* pk1 = &sk[16 + c0][kb16];   // cols kc+16..kc+31
      const _Float16* pv0 = &sv[c0][kb16];        // B[kpos][d], d cols 0..15
      const _Float16* pv1 = &sv[16 + c0][kb16];   // d cols 16..31
#pragma unroll
      for (int e = 0; e < 16; ++e) {
        kB0[e] = pk0[e];
        kB1[e] = pk1[e];
        vB0[e] = pv0[e];
        vB1[e] = pv1[e];
      }
    }
    v8f z = {};
    v8f S0 = wmma_f16(qA, kB0, z);
    v8f S1 = wmma_f16(qA, kB1, z);
    float p0[8], p1[8];
#pragma unroll
    for (int e = 0; e < 8; ++e) {
      int qr = qt * 16 + rb + e;
      float s0 = fmaf(S0[e], scale, (float)cp[(size_t)qr * KLEN + kc + c0]);
      float s1 = fmaf(S1[e], scale, (float)cp[(size_t)qr * KLEN + kc + 16 + c0]);
      float tmax = fmaxf(s0, s1);
#pragma unroll
      for (int off = 1; off < 16; off <<= 1) tmax = fmaxf(tmax, __shfl_xor(tmax, off, 16));
      float mn = fmaxf(mi[e], tmax);
      float alpha = __expf(mi[e] - mn);
      mi[e] = mn;
      p0[e] = __expf(s0 - mn);
      p1[e] = __expf(s1 - mn);
      float rs = p0[e] + p1[e];
#pragma unroll
      for (int off = 1; off < 16; off <<= 1) rs += __shfl_xor(rs, off, 16);
      li[e] = li[e] * alpha + rs;
      O0[e] *= alpha;
      O1[e] *= alpha;
    }
    // transpose P (C layout -> A layout) through per-wave LDS tile
#pragma unroll
    for (int e = 0; e < 8; ++e) {
      pbuf[wid][rb + e][c0] = (_Float16)p0[e];
      pbuf[wid][rb + e][16 + c0] = (_Float16)p1[e];
    }
    __builtin_amdgcn_wave_barrier();
    v16h pA;
    {
      const _Float16* pr = &pbuf[wid][lane & 15][(lane & 16) ? 16 : 0];
#pragma unroll
      for (int e = 0; e < 8; ++e) { pA[e] = pr[e]; pA[8 + e] = pr[8 + e]; }
    }
    __builtin_amdgcn_wave_barrier();
    O0 = wmma_f16(pA, vB0, O0);
    O1 = wmma_f16(pA, vB1, O1);
    __syncthreads();
  }
#pragma unroll
  for (int e = 0; e < 8; ++e) {
    float inv = 1.0f / li[e];
    O0[e] *= inv;
    O1[e] *= inv;
  }
  // att[b, q, h*32 + d]  (matches transpose(0,2,1,3).reshape)
#pragma unroll
  for (int e = 0; e < 8; ++e) {
    int qr = qt * 16 + rb + e;
    size_t base = (((size_t)b * QLEN + qr) * HEADS + h) * HD;
    att[base + c0] = (_Float16)O0[e];
    att[base + 16 + c0] = (_Float16)O1[e];
  }
}

// ---------------- Kernel 5: output projection ----------------
// out[m,n] = att(f16)[m,:] @ Wo[n,:] + bo[n], M=512 N=256 K=256, f32 out.
__global__ void oproj_kernel(const half_t* __restrict__ att, const float* __restrict__ Wo,
                             const float* __restrict__ bo, float* __restrict__ out) {
  int wid = threadIdx.x >> 5, lane = threadIdx.x & 31;
  int tile = blockIdx.x * 4 + wid;  // 512 tiles
  int mt = tile >> 4, nt = tile & 15;
  v8f acc = {};
#pragma unroll
  for (int kc = 0; kc < DIM; kc += 32) {
    v16h a = loadA_f16(att, DIM, mt * 16, kc);
    v16h w = loadB_NK_f32(Wo, DIM, nt * 16, kc);
    acc = wmma_f16(a, w, acc);
  }
  int n = nt * 16 + (lane & 15);
  float bias = bo[n];
  int rb = (lane & 16) ? 8 : 0;
#pragma unroll
  for (int e = 0; e < 8; ++e) {
    int m = mt * 16 + rb + e;
    out[(size_t)m * DIM + n] = acc[e] + bias;
  }
}

extern "C" void kernel_launch(void* const* d_in, const int* in_sizes, int n_in,
                              void* d_out, int out_size, void* d_ws, size_t ws_size,
                              hipStream_t stream) {
  const float* entities = (const float*)d_in[0];
  const float* images   = (const float*)d_in[1];
  const float* rd       = (const float*)d_in[2];
  const float* Wq       = (const float*)d_in[3];
  const float* bq       = (const float*)d_in[4];
  const float* Wkv      = (const float*)d_in[5];
  const float* bkv      = (const float*)d_in[6];
  const float* W1       = (const float*)d_in[7];
  const float* b1       = (const float*)d_in[8];
  const float* W2       = (const float*)d_in[9];
  const float* b2       = (const float*)d_in[10];
  const float* Wo       = (const float*)d_in[11];
  const float* bo       = (const float*)d_in[12];

  half_t* ws    = (half_t*)d_ws;
  half_t* q16   = ws;                      // 131072
  half_t* k16   = q16 + 131072;            // 1048576
  half_t* vT16  = k16 + 1048576;           // 1048576 (transposed [b,h,d,kpos])
  half_t* att16 = vT16 + 1048576;          // 131072
  half_t* cpb16 = att16 + 131072;          // 8388608  (~20.5 MB total)

  proj_q_kernel<<<128, 128, 0, stream>>>(entities, Wq, bq, q16);
  proj_kv_kernel<<<2048, 128, 0, stream>>>(images, Wkv, bkv, k16, vT16);
  cpb_kernel<<<8192, 256, 0, stream>>>(rd, W1, b1, W2, b2, cpb16);
  attn_kernel<<<64, 128, 0, stream>>>(q16, k16, vT16, cpb16, att16);
  oproj_kernel<<<128, 128, 0, stream>>>(att16, Wo, bo, (float*)d_out);
}
